// NequIP_67001489817864
// MI455X (gfx1250) — matrix operand order repaired and checked by reading, not compiled, not run on hardware
//
#include <hip/hip_runtime.h>

#define DEV __device__ __forceinline__

typedef __attribute__((ext_vector_type(16))) __bf16 v16bf;
typedef __attribute__((ext_vector_type(2)))  __bf16 v2bf;
typedef __attribute__((ext_vector_type(8)))  float  v8f;

constexpr int   GIN  = 92;   // conv output features (GATE_IN)
constexpr int   GOUT = 76;   // gated features (GATE_OUT)
constexpr float CUT  = 5.0f;

// Pack two f32 -> bf16x2 dword (round-to-nearest-even). Staging-time only.
DEV unsigned pk_bf16(float lo, float hi) {
  union { float f; unsigned u; } a, b;
  a.f = lo; b.f = hi;
  unsigned lr = (a.u + 0x7FFFu + ((a.u >> 16) & 1u)) >> 16;
  unsigned hr = (b.u + 0x7FFFu + ((b.u >> 16) & 1u)) >> 16;
  return (hr << 16) | (lr & 0xFFFFu);
}

// Packed bf16x2 multiply (targets v_pk_mul_bf16 on gfx1250).
DEV unsigned mul_pk_bf16(unsigned a, unsigned b) {
  v2bf x = __builtin_bit_cast(v2bf, a);
  v2bf y = __builtin_bit_cast(v2bf, b);
  v2bf r = x * y;
  return __builtin_bit_cast(unsigned, r);
}

// ---------------------------------------------------------------- utilities
__global__ void zero_kernel(float* __restrict__ p, long n) {
  long i = (long)blockIdx.x * blockDim.x + threadIdx.x;
  if (i < n) p[i] = 0.f;
}

__global__ void embed_kernel(const int* __restrict__ ti, const float* __restrict__ tbl,
                             float* __restrict__ te, int N) {
  int n = blockIdx.x * blockDim.x + threadIdx.x;
  if (n >= N) return;
  int t = ti[n];
#pragma unroll
  for (int k = 0; k < 4; ++k) te[n * 4 + k] = tbl[t * 4 + k];
}

// -------------------------------------------------- edge features: shw1/2/3
DEV void mlp_shw(const float* ee, const float* sh,
                 const float* __restrict__ w1, const float* __restrict__ w2,
                 float* out) {
  float h[10];
#pragma unroll
  for (int j = 0; j < 10; ++j) {
    float a = 0.f;
#pragma unroll
    for (int k = 0; k < 18; ++k) a += ee[k] * w1[k * 10 + j];
    h[j] = fmaxf(a, 0.f);
  }
#pragma unroll
  for (int s = 0; s < 9; ++s) {
    float a = 0.f;
#pragma unroll
    for (int j = 0; j < 10; ++j) a += h[j] * w2[j * 9 + s];
    out[s] = sh[s] * a;
  }
}

__global__ void edge_feat_kernel(const float* __restrict__ pos,
                                 const int* __restrict__ esrc, const int* __restrict__ edst,
                                 const float* __restrict__ te,
                                 const float* __restrict__ w1a, const float* __restrict__ w2a,
                                 const float* __restrict__ w1b, const float* __restrict__ w2b,
                                 const float* __restrict__ w1c, const float* __restrict__ w2c,
                                 float* __restrict__ shw1, float* __restrict__ shw2,
                                 float* __restrict__ shw3, int E) {
  int e = blockIdx.x * blockDim.x + threadIdx.x;
  if (e >= E) return;
  int s = esrc[e], d = edst[e];
  float ex = pos[d * 3 + 0] - pos[s * 3 + 0];
  float ey = pos[d * 3 + 1] - pos[s * 3 + 1];
  float ez = pos[d * 3 + 2] - pos[s * 3 + 2];
  float r = sqrtf(ex * ex + ey * ey + ez * ez);
  float rinv = 1.0f / fmaxf(r, 1e-9f);
  float ux = ex * rinv, uy = ey * rinv, uz = ez * rinv;
  const float c1 = 1.7320508075688772f;   // sqrt(3)
  const float c2 = 3.872983346207417f;    // sqrt(15)
  const float c2b = 1.118033988749895f;   // sqrt(5)/2
  float sh[9] = {1.f, c1 * uy, c1 * uz, c1 * ux,
                 c2 * ux * uy, c2 * uy * uz, c2b * (3.f * uz * uz - 1.f),
                 c2 * ux * uz, 0.5f * c2 * (ux * ux - uy * uy)};
  // radial Bessel basis with polynomial envelope (p=6)
  float xq = r / CUT;
  float x2 = xq * xq, x3 = x2 * xq;
  float x6 = x3 * x3, x7 = x6 * xq, x8 = x7 * xq;
  float env = 1.f - 28.f * x6 + 48.f * x7 - 21.f * x8;
  float pref = 0.6324555320336759f * rinv * env;  // sqrt(2/5)
  float ee[18];
#pragma unroll
  for (int k = 0; k < 4; ++k) ee[k] = te[d * 4 + k];
#pragma unroll
  for (int k = 0; k < 4; ++k) ee[4 + k] = te[s * 4 + k];
  // sin(n*pi*x) via Chebyshev recurrence: 1 sinf + 1 cosf total
  {
    float th = 3.14159265358979323846f * xq;
    float s1 = sinf(th), cth = 2.f * cosf(th);
    float sm1 = 0.f, sn = s1;
#pragma unroll
    for (int n = 1; n <= 10; ++n) {
      ee[7 + n] = pref * sn;
      float snext = cth * sn - sm1;
      sm1 = sn;
      sn = snext;
    }
  }

  float o[9];
  mlp_shw(ee, sh, w1a, w2a, o);
#pragma unroll
  for (int k = 0; k < 9; ++k) shw1[(long)e * 9 + k] = o[k];
  mlp_shw(ee, sh, w1b, w2b, o);
#pragma unroll
  for (int k = 0; k < 9; ++k) shw2[(long)e * 9 + k] = o[k];
  mlp_shw(ee, sh, w1c, w2c, o);
#pragma unroll
  for (int k = 0; k < 9; ++k) shw3[(long)e * 9 + k] = o[k];
}

// -------------------------------------------------- WMMA conv (conv1/conv2)
// msg[e,:] = z[e,:] @ Bflat, z[e, s*F_IN+i] = shw[e,s]*x[src[e],i],
// Bflat[(s*F_IN+i), n] = tp[s][i][n].
// Each wave owns TWO 16-edge M-tiles (32 edges); x/shw staged in LDS as
// packed bf16 pairs, so the k-loop A build is 1 ds_load_b32 + 1 v_pk_mul_bf16
// per dword. k-loop fully unrolled so shw indices are compile-time registers.
template <int F_IN>
__attribute__((amdgpu_waves_per_eu(1, 2)))
__launch_bounds__(256)
__global__ void conv_wmma_kernel(const float* __restrict__ x, const float* __restrict__ shw,
                                 const int* __restrict__ esrc, const int* __restrict__ edst,
                                 const float* __restrict__ tp, float* __restrict__ acc, int E) {
  constexpr int KTOT = 9 * F_IN;
  constexpr int NK = (KTOT + 31) / 32;   // k-steps of 32 (bf16 WMMA K)
  constexpr int NT = 6;                  // 92 -> 6 N-tiles of 16
  constexpr int NW = 8;                  // waves per block
  constexpr int NQ = F_IN / 4;           // float4 per x row (F_IN % 4 == 0)
  constexpr int PF = F_IN / 2;           // bf16x2 dwords per x row

  // B in WMMA fragment layout: frag = (ks*NT+nt), 32 lanes x 8 dwords (bf16x2).
  __shared__ __align__(16) unsigned bfrag[NK * NT * 256];
  __shared__ __align__(16) unsigned xgb[NW][32 * PF];  // gathered x rows, bf16 pairs
  __shared__ unsigned shb[NW][32 * 9];                 // shw duplicated bf16x2
  __shared__ int dstl[NW][32];
  __shared__ int srcl[NW][32];

  const int tid = threadIdx.x;
  // ---- stage B once per block (tp is L2-resident; amortized over many tiles)
  for (int idx = tid; idx < NK * NT * 256; idx += 256) {
    int f = idx >> 8;
    int rr = idx & 255;
    int l = rr >> 3, v = rr & 7;
    int ks = f / NT, nt = f - ks * NT;
    int n = nt * 16 + (l & 15);
    // B layout: lanes 0-15 hold K in {0..7,16..23}; lanes 16-31 hold K in {8..15,24..31}
    int k = ks * 32 + ((l >> 4) << 3) + ((v & 4) << 2) + ((v & 3) << 1);
    float f0 = 0.f, f1 = 0.f;
    if (k < KTOT && n < GIN) {
      int s = k / F_IN, i = k - s * F_IN;         // F_IN even -> pair stays in-bounds
      const float* tpp = tp + (s * F_IN + i) * GIN + n;
      f0 = tpp[0];
      f1 = tpp[GIN];
    }
    bfrag[idx] = pk_bf16(f0, f1);
  }
  __syncthreads();

  const int wave = tid >> 5, lane = tid & 31;
  const int m16 = lane & 15, kset = lane >> 4;
  const int ntile = (E + 31) >> 5;
  const int tstride = gridDim.x * NW;
  for (int t = blockIdx.x * NW + wave; t < ntile; t += tstride) {
    const int e0 = t << 5;
    // prefetch next tile's edge metadata
    if (t + tstride < ntile) {
      int en = (t + tstride) << 5;
      __builtin_prefetch(&esrc[en], 0, 1);
      __builtin_prefetch(&edst[en], 0, 1);
      __builtin_prefetch(&shw[(long)en * 9], 0, 1);
    }
    // ---- stage per-tile edge data (wave-private LDS region)
    {
      int e = e0 + lane;
      bool ok = e < E;
      srcl[wave][lane] = ok ? esrc[e] : 0;
      dstl[wave][lane] = ok ? edst[e] : -1;
#pragma unroll
      for (int s = 0; s < 9; ++s) {
        float w = ok ? shw[(long)e * 9 + s] : 0.f;
        shb[wave][lane * 9 + s] = pk_bf16(w, w);
      }
    }
    asm volatile("s_wait_dscnt 0" ::: "memory");
    // gathered x rows -> bf16 pairs, float4-wide reads
    for (int idx = lane; idx < 32 * NQ; idx += 32) {
      int m = idx / NQ, q = idx - m * NQ;
      float4 v = make_float4(0.f, 0.f, 0.f, 0.f);
      if (e0 + m < E)
        v = ((const float4*)(x + (long)srcl[wave][m] * F_IN))[q];
      uint2 st;
      st.x = pk_bf16(v.x, v.y);
      st.y = pk_bf16(v.z, v.w);
      ((uint2*)&xgb[wave][0])[idx] = st;
    }
    asm volatile("s_wait_dscnt 0" ::: "memory");

    // per-lane shw bf16x2 registers for its two rows (constant-indexed)
    unsigned sh0[9], sh1[9];
#pragma unroll
    for (int s = 0; s < 9; ++s) {
      sh0[s] = shb[wave][m16 * 9 + s];
      sh1[s] = shb[wave][(16 + m16) * 9 + s];
    }
    const unsigned* xr0 = &xgb[wave][m16 * PF];
    const unsigned* xr1 = &xgb[wave][(16 + m16) * PF];

    v8f c0[NT] = {};
    v8f c1[NT] = {};
#pragma unroll
    for (int ks = 0; ks < NK; ++ks) {
      union AU { unsigned u[8]; v16bf v; } A0, A1;
#pragma unroll
      for (int v = 0; v < 8; ++v) {
        // A layout: lanes 0-15 K in {0..7,16..23}; lanes 16-31 K in {8..15,24..31}
        const int kA = ks * 32 + ((v & 4) << 2) + ((v & 3) << 1);  // kset==0
        const int kB = kA + 8;                                     // kset==1
        const bool vA = kA < KTOT, vB = kB < KTOT;
        const int sA = vA ? (kA / F_IN) : 0;
        const int sB = vB ? (kB / F_IN) : 0;
        const int pA = vA ? ((kA - sA * F_IN) >> 1) : 0;
        const int pB = vB ? ((kB - sB * F_IN) >> 1) : 0;
        int p = kset ? pB : pA;  // select between two compile-time constants
        unsigned x0 = xr0[p];
        unsigned x1 = xr1[p];
        unsigned s0 = (sA == sB) ? sh0[sA] : (kset ? sh0[sB] : sh0[sA]);
        unsigned s1 = (sA == sB) ? sh1[sA] : (kset ? sh1[sB] : sh1[sA]);
        unsigned r0 = mul_pk_bf16(x0, s0);
        unsigned r1 = mul_pk_bf16(x1, s1);
        if (vA && vB) {
          A0.u[v] = r0;
          A1.u[v] = r1;
        } else if (!vA && !vB) {
          A0.u[v] = 0u;
          A1.u[v] = 0u;
        } else {  // mixed validity across lane halves
          unsigned okm = (kset ? vB : vA) ? 0xFFFFFFFFu : 0u;
          A0.u[v] = r0 & okm;
          A1.u[v] = r1 & okm;
        }
      }
#pragma unroll
      for (int nt = 0; nt < NT; ++nt) {
        union { uint4 q[2]; v16bf v; } B;
        const uint4* bp = (const uint4*)&bfrag[(ks * NT + nt) * 256 + lane * 8];
        B.q[0] = bp[0];
        B.q[1] = bp[1];
        c0[nt] = __builtin_amdgcn_wmma_f32_16x16x32_bf16(
            false, A0.v, false, B.v, (short)0, c0[nt], false, false);
        c1[nt] = __builtin_amdgcn_wmma_f32_16x16x32_bf16(
            false, A1.v, false, B.v, (short)0, c1[nt], false, false);
      }
    }
    // ---- segment-sum scatter (fold 1/sqrt(AVG_DEG)=0.25 here)
#pragma unroll
    for (int nt = 0; nt < NT; ++nt) {
      int col = nt * 16 + m16;
      if (col < GIN) {
#pragma unroll
        for (int j = 0; j < 8; ++j) {
          int mrow = j + (kset << 3);  // C layout: M = vgpr + 8*(lane>=16)
          int d0 = dstl[wave][mrow];
          if (d0 >= 0)
            unsafeAtomicAdd(&acc[(long)d0 * GIN + col], c0[nt][j] * 0.25f);
          int d1 = dstl[wave][16 + mrow];
          if (d1 >= 0)
            unsafeAtomicAdd(&acc[(long)d1 * GIN + col], c1[nt][j] * 0.25f);
        }
      }
    }
  }
}

// ---------------------------------------------------------------- gate (+resid)
// Streams outputs directly (no y[] register array -> no spill pressure).
__global__ void gate_kernel(const float* __restrict__ acc, const float* __restrict__ resid,
                            float* __restrict__ xout, float* __restrict__ accz,
                            int N, int mode) {
  int n = blockIdx.x * blockDim.x + threadIdx.x;
  if (n >= N) return;
  const float* in = acc + (long)n * GIN;
  float* yo = xout + (long)n * GOUT;
  const float* rs = resid + (mode == 1 ? (long)n * 4 : (long)n * GOUT);
  float ga[16];
#pragma unroll
  for (int k = 0; k < 4; ++k) ga[k] = fmaxf(in[12 + k], 0.f);
#pragma unroll
  for (int k = 0; k < 4; ++k) ga[4 + k] = tanhf(in[16 + k]);
#pragma unroll
  for (int k = 0; k < 4; ++k) ga[8 + k] = fmaxf(in[20 + k], 0.f);
#pragma unroll
  for (int k = 0; k < 4; ++k) ga[12 + k] = tanhf(in[24 + k]);
#pragma unroll
  for (int k = 0; k < 4; ++k) {
    float r = (mode == 1) ? rs[k] : rs[k];
    yo[k] = fmaxf(in[k], 0.f) + r;
  }
#pragma unroll
  for (int k = 4; k < 12; ++k)
    yo[k] = fabsf(in[k]) + ((mode == 1) ? 0.f : rs[k]);
#pragma unroll
  for (int j = 0; j < 24; ++j)
    yo[12 + j] = in[28 + j] * ga[j / 3] + ((mode == 1) ? 0.f : rs[12 + j]);
#pragma unroll
  for (int j = 0; j < 40; ++j)
    yo[36 + j] = in[52 + j] * ga[8 + j / 5] + ((mode == 1) ? 0.f : rs[36 + j]);
  float* az = accz + (long)n * GIN;
#pragma unroll
  for (int k = 0; k < GIN; ++k) az[k] = 0.f;  // re-zero accumulator for next conv
}

// ------------------------------------------- conv3 (F_out=1) + global reduce
// sum over nodes of segment_sum == sum over edges of msg, so no scatter at all.
// Rolled loop to keep live ranges small (previous version spilled 3KB/lane).
__global__ void conv3_kernel(const float* __restrict__ x2, const float* __restrict__ shw3,
                             const int* __restrict__ esrc, const float* __restrict__ tp3,
                             float* __restrict__ out, int E) {
  __shared__ float tpl[684];  // tp3 (9,76,1) flattened
  for (int i = threadIdx.x; i < 684; i += blockDim.x) tpl[i] = tp3[i];
  __syncthreads();
  float msg = 0.f;
  for (int e = blockIdx.x * blockDim.x + threadIdx.x; e < E; e += gridDim.x * blockDim.x) {
    float w[9];
#pragma unroll
    for (int s = 0; s < 9; ++s) w[s] = shw3[(long)e * 9 + s];
    const float4* xp = (const float4*)(x2 + (long)esrc[e] * GOUT);
    float a = 0.f;
#pragma unroll 1
    for (int q = 0; q < GOUT / 4; ++q) {
      float4 xv = xp[q];
      float c0 = 0.f, c1 = 0.f, c2 = 0.f, c3 = 0.f;
#pragma unroll
      for (int s = 0; s < 9; ++s) {
        const float* tb = &tpl[s * GOUT + q * 4];
        c0 += w[s] * tb[0];
        c1 += w[s] * tb[1];
        c2 += w[s] * tb[2];
        c3 += w[s] * tb[3];
      }
      a += xv.x * c0 + xv.y * c1 + xv.z * c2 + xv.w * c3;
    }
    msg += a;
  }
  // wave32 shuffle reduction
  for (int off = 16; off; off >>= 1) msg += __shfl_down(msg, off, 32);
  if ((threadIdx.x & 31) == 0) unsafeAtomicAdd(out, msg * 0.25f);
}

// ----------------------------------------------------------------- launcher
extern "C" void kernel_launch(void* const* d_in, const int* in_sizes, int n_in,
                              void* d_out, int out_size, void* d_ws, size_t ws_size,
                              hipStream_t stream) {
  (void)n_in; (void)out_size; (void)ws_size;
  const int*   ti   = (const int*)d_in[0];
  const float* pos  = (const float*)d_in[1];
  const int*   esrc = (const int*)d_in[2];
  const int*   edst = (const int*)d_in[3];
  const float* tbl  = (const float*)d_in[4];
  const float* m1w1 = (const float*)d_in[5];
  const float* m1w2 = (const float*)d_in[6];
  const float* tp1  = (const float*)d_in[7];
  const float* m2w1 = (const float*)d_in[8];
  const float* m2w2 = (const float*)d_in[9];
  const float* tp2  = (const float*)d_in[10];
  const float* m3w1 = (const float*)d_in[11];
  const float* m3w2 = (const float*)d_in[12];
  const float* tp3  = (const float*)d_in[13];
  const int N = in_sizes[0];
  const int E = in_sizes[2];

  char* ws = (char*)d_ws;
  size_t off = 0;
  auto wsalloc = [&](size_t bytes) {
    void* p = ws + off;
    off += (bytes + 255) & ~(size_t)255;
    return p;
  };
  float* te   = (float*)wsalloc((size_t)N * 4 * 4);
  float* x1   = (float*)wsalloc((size_t)N * GOUT * 4);
  float* x2   = (float*)wsalloc((size_t)N * GOUT * 4);
  float* acc  = (float*)wsalloc((size_t)N * GIN * 4);
  float* shw1 = (float*)wsalloc((size_t)E * 9 * 4);
  float* shw2 = (float*)wsalloc((size_t)E * 9 * 4);
  float* shw3 = (float*)wsalloc((size_t)E * 9 * 4);
  float* outp = (float*)d_out;

  dim3 b256(256);
  long accn = (long)N * GIN;
  zero_kernel<<<(unsigned)((accn + 255) / 256), b256, 0, stream>>>(acc, accn);
  zero_kernel<<<1, 32, 0, stream>>>(outp, 1);
  embed_kernel<<<(N + 255) / 256, b256, 0, stream>>>(ti, tbl, te, N);
  edge_feat_kernel<<<(E + 255) / 256, b256, 0, stream>>>(
      pos, esrc, edst, te, m1w1, m1w2, m2w1, m2w2, m3w1, m3w2, shw1, shw2, shw3, E);
  conv_wmma_kernel<4><<<768, b256, 0, stream>>>(te, shw1, esrc, edst, tp1, acc, E);
  gate_kernel<<<(N + 255) / 256, b256, 0, stream>>>(acc, te, x1, acc, N, 1);
  conv_wmma_kernel<76><<<768, b256, 0, stream>>>(x1, shw2, esrc, edst, tp2, acc, E);
  gate_kernel<<<(N + 255) / 256, b256, 0, stream>>>(acc, x1, x2, acc, N, 2);
  conv3_kernel<<<2048, b256, 0, stream>>>(x2, shw3, esrc, tp3, outp, E);
}
